// MultiHeadAttention_13692355740195
// MI455X (gfx1250) — matrix operand order repaired and checked by reading, not compiled, not run on hardware
//
#include <hip/hip_runtime.h>

#define NTOK    32
#define DMODEL  256
#define NTYPES  4
#define NHEAD   8
#define BATCH   4096
#define GB      8                 // batches per block
#define NBLK    (BATCH / GB)      // 512
#define NROW    (GB * NTOK)       // 256 token rows per block
#define R16     264               // padded f16 row stride (bank-conflict-free)

typedef __attribute__((ext_vector_type(16))) _Float16 v16h;
typedef __attribute__((ext_vector_type(8)))  _Float16 v8h;
typedef __attribute__((ext_vector_type(4)))  _Float16 v4h;
typedef __attribute__((ext_vector_type(8)))  float    v8f;

// Packed B-fragment store: [e][nt(16)][ks(8)] fragments of 512 halves,
// internal order = lane*16 + h  => one coalesced 32B/lane load per wave.
#define FRAG_HALVES (NTYPES * 16 * 8 * 512)   // 262144 per weight matrix

// ---------------------------------------------------------------------------
// Prep: fp32 weights -> f16 WMMA B-fragments.
// B is K=32 x N=16; n=lane&15; half h<8 -> k=8*(lane>>4)+h; half h>=8 ->
// k=16+8*(lane>>4)+(h-8).
// ---------------------------------------------------------------------------
__global__ void pack_w_kernel(const float* __restrict__ W, _Float16* __restrict__ F) {
  const int idx  = blockIdx.x * 256 + threadIdx.x;
  const int h    = idx & 15;
  const int lane = (idx >> 4) & 31;
  const int fi   = idx >> 9;
  const int ks   = fi & 7;
  const int nt   = (fi >> 3) & 15;
  const int e    = fi >> 7;
  const int n    = nt * 16 + (lane & 15);
  const int k    = ks * 32 + ((lane >> 4) << 3) + (h & 7) + ((h >> 3) << 4);
  F[idx] = (_Float16)W[e * (DMODEL * DMODEL) + k * DMODEL + n];
}

// ---------------------------------------------------------------------------
// Dynamic LDS layout (bytes)
// ---------------------------------------------------------------------------
#define OFF_STAGE 0        // 256 x R16 f16 : staging q -> k -> v
#define OFF_MAT   135168   // 256 x R16 f16 : Q(f16) then V(f16)
#define OFF_SC    270336   // 8 x 8 x 32 f32 scores/attn
#define OFF_FCIN  278528   // 16 x R16 f16 : FC A-matrix (8 valid rows)
#define OFF_PROJ  286976   // 8 x 256 f32
#define OFF_ET    295168   // 256 int
#define OFF_PRI   296192   // 256 f32
#define OFF_MSK   297216   // 256 int
#define OFF_US    298240   // 16 int
#define SMEM_BYTES 298304  // < 320KB per-workgroup limit

__device__ __forceinline__ void stage_f16(const float* __restrict__ src,
                                          _Float16* __restrict__ dst, int t) {
  for (int j = 0; j < 64; ++j) {
    const int idx = j * 1024 + t * 4;          // 0..65535, coalesced float4
    const int row = idx >> 8, c = idx & 255;
    float4 f = *(const float4*)(src + idx);
    v4h hv = {(_Float16)f.x, (_Float16)f.y, (_Float16)f.z, (_Float16)f.w};
    *(v4h*)(dst + row * R16 + c) = hv;
  }
}

// Prefetch one 256KB stage (2048 x 128B lines) into cache while WMMA runs:
// emits global_prefetch_b8 (gfx1250 prefetch path), no counters touched.
__device__ __forceinline__ void prefetch_stage(const float* __restrict__ src, int t) {
  for (int j = 0; j < 8; ++j)
    __builtin_prefetch(src + (size_t)(j * 256 + t) * 32, 0, 3);
}

// A-fragment (16x32 f16): lane<16 row=M=lane&15, halves 0..7 -> K=kb..kb+7,
// halves 8..15 -> K=kb+16..+23; lanes 16-31 same rows, K offset +8.
__device__ __forceinline__ v16h load_afrag(const _Float16* __restrict__ rowBase, int kb) {
  union { v16h v; v8h h2[2]; } au;
  au.h2[0] = *(const v8h*)(rowBase + kb);
  au.h2[1] = *(const v8h*)(rowBase + kb + 16);
  return au.v;
}

// One M-pass of a type-masked 256x256 projection: wave owns 4 M-tiles x 2 N-tiles.
__device__ __forceinline__ void pass_accumulate(const _Float16* __restrict__ stageBuf,
                                                const _Float16* __restrict__ wfrag,
                                                const int* __restrict__ et,
                                                int p, int lane, int wave,
                                                v8f acc[4][2]) {
  const int hl = lane & 15, hi = lane >> 4;
  for (int e = 0; e < NTYPES; ++e) {
    for (int ks = 0; ks < 8; ++ks) {
      const int kb = ks * 32 + hi * 8;
      v16h a[4];
#pragma unroll
      for (int i = 0; i < 4; ++i) {
        const int row = (p * 4 + i) * 16 + hl;
        const _Float16 s = (et[row] == e) ? (_Float16)1.0f : (_Float16)0.0f;
        a[i] = load_afrag(stageBuf + row * R16, kb) * s;
      }
#pragma unroll
      for (int j = 0; j < 2; ++j) {
        const int nt = wave * 2 + j;
        const v16h b = *(const v16h*)(wfrag + (((e * 16 + nt) * 8 + ks) << 9) + lane * 16);
#pragma unroll
        for (int i = 0; i < 4; ++i)
          acc[i][j] = __builtin_amdgcn_wmma_f32_16x16x32_f16(
              false, a[i], false, b, (short)0, acc[i][j], false, false);
      }
    }
  }
}

// Full projection (Q or V): accumulate and store result as f16 into matOut.
__device__ __forceinline__ void proj_phase(const _Float16* __restrict__ stageBuf,
                                           const _Float16* __restrict__ wfrag,
                                           _Float16* __restrict__ matOut,
                                           const int* __restrict__ et,
                                           int lane, int wave) {
  const int hl = lane & 15, hi = lane >> 4;
  for (int p = 0; p < 4; ++p) {
    v8f acc[4][2] = {};
    pass_accumulate(stageBuf, wfrag, et, p, lane, wave, acc);
#pragma unroll
    for (int i = 0; i < 4; ++i) {
      const int rbase = (p * 4 + i) * 16 + hi * 8;    // C/D: lane holds col, r = row
#pragma unroll
      for (int j = 0; j < 2; ++j) {
        const int col = (wave * 2 + j) * 16 + hl;
#pragma unroll
        for (int r = 0; r < 8; ++r)
          matOut[(rbase + r) * R16 + col] = (_Float16)acc[i][j][r];
      }
    }
  }
}

// K projection fused with diagonal score: wave w's 2 N-tiles == head w's 32 cols,
// so score[token] reduces fully inside the wave (products + 4 shfl_xor steps).
__device__ __forceinline__ void kscore_phase(const _Float16* __restrict__ stageBuf,
                                             const _Float16* __restrict__ wfrag,
                                             const _Float16* __restrict__ qmat,
                                             float* __restrict__ sc,
                                             const int* __restrict__ et,
                                             const float* __restrict__ pri,
                                             const int* __restrict__ msk,
                                             int lane, int wave) {
  const int hl = lane & 15, hi = lane >> 4;
  for (int p = 0; p < 4; ++p) {
    v8f acc[4][2] = {};
    pass_accumulate(stageBuf, wfrag, et, p, lane, wave, acc);
#pragma unroll
    for (int i = 0; i < 4; ++i) {
      const int rbase = (p * 4 + i) * 16 + hi * 8;
      float part[8];
#pragma unroll
      for (int r = 0; r < 8; ++r) {
        const int m = rbase + r;
        const float q0 = (float)qmat[m * R16 + (wave * 2 + 0) * 16 + hl];
        const float q1 = (float)qmat[m * R16 + (wave * 2 + 1) * 16 + hl];
        part[r] = acc[i][0][r] * q0 + acc[i][1][r] * q1;
      }
#pragma unroll
      for (int r = 0; r < 8; ++r) {
        float s = part[r];
        s += __shfl_xor(s, 1); s += __shfl_xor(s, 2);
        s += __shfl_xor(s, 4); s += __shfl_xor(s, 8);
        if (hl == 0) {                         // one lane per 16-lane group
          const int m = rbase + r;             // token row
          float val = s * pri[m] * 0.17677669529663688f;   // 1/sqrt(32)
          if (msk[m]) val = -1.0e10f;
          sc[(m >> 5) * 256 + wave * 32 + (m & 31)] = val; // [g][h=wave][n]
        }
      }
    }
  }
}

// ---------------------------------------------------------------------------
// Fused kernel: 1 block = 8 batches, 256 threads = 8 wave32s.
// ---------------------------------------------------------------------------
__global__ __launch_bounds__(256) void mha_fused_kernel(
    const float* __restrict__ q, const float* __restrict__ k, const float* __restrict__ v,
    const _Float16* __restrict__ fWq, const _Float16* __restrict__ fWk,
    const _Float16* __restrict__ fWv, const _Float16* __restrict__ fFc,
    const float* __restrict__ rel_pri, const float* __restrict__ fcb,
    const float* __restrict__ gamma_, const float* __restrict__ beta_,
    const int* __restrict__ etype, const int* __restrict__ utype,
    const unsigned char* __restrict__ mask, float* __restrict__ out) {
  extern __shared__ __align__(16) char smem[];
  _Float16* stg  = (_Float16*)(smem + OFF_STAGE);
  _Float16* mat  = (_Float16*)(smem + OFF_MAT);
  float*    sc   = (float*)(smem + OFF_SC);
  _Float16* fcin = (_Float16*)(smem + OFF_FCIN);
  float*    proj = (float*)(smem + OFF_PROJ);
  int*      et   = (int*)(smem + OFF_ET);
  float*    pri  = (float*)(smem + OFF_PRI);
  int*      msk  = (int*)(smem + OFF_MSK);
  int*      us   = (int*)(smem + OFF_US);

  const int blk = blockIdx.x, t = threadIdx.x;
  const int lane = t & 31, wave = t >> 5;
  const int hl = lane & 15, hi = lane >> 4;

  const float* qsrc = q + (size_t)blk * (NROW * DMODEL);
  const float* ksrc = k + (size_t)blk * (NROW * DMODEL);
  const float* vsrc = v + (size_t)blk * (NROW * DMODEL);

  // ---- metadata + stage q ----
  {
    const int e = etype[blk * NROW + t];               // token t of this block
    et[t]  = e;
    msk[t] = (int)mask[blk * NROW + t];
    pri[t] = rel_pri[utype[blk * GB + (t >> 5)] * NTYPES + e];
  }
  if (t < 16) us[t] = (t < GB) ? utype[blk * GB + t] : -1;
  stage_f16(qsrc, stg, t);
  prefetch_stage(ksrc, t);                 // overlap k's HBM fetch with Q WMMA
  __syncthreads();

  // ---- Q projection -> mat (f16) ----
  proj_phase(stg, fWq, mat, et, lane, wave);
  __syncthreads();

  // ---- stage k, K projection fused with diagonal scores ----
  stage_f16(ksrc, stg, t);
  prefetch_stage(vsrc, t);                 // overlap v's HBM fetch with K WMMA
  __syncthreads();
  kscore_phase(stg, fWk, mat, sc, et, pri, msk, lane, wave);
  __syncthreads();

  // ---- stage v, V projection -> mat (overwrites Q) ----
  stage_f16(vsrc, stg, t);
  __syncthreads();
  proj_phase(stg, fWv, mat, et, lane, wave);

  // ---- softmax: wave w = batch g=w; lane = token n ----
  {
    const int g = wave;
    for (int h = 0; h < NHEAD; ++h) {
      float sval = sc[g * 256 + h * 32 + lane];
      float mx = sval;
      for (int off = 16; off > 0; off >>= 1) mx = fmaxf(mx, __shfl_xor(mx, off));
      float ex = __expf(sval - mx);
      float sm = ex;
      for (int off = 16; off > 0; off >>= 1) sm += __shfl_xor(sm, off);
      const float a = ex / sm;
      sc[g * 256 + h * 32 + lane] = a;
      // attn_flat[(h*B + b)*N + n]
      out[(size_t)BATCH * DMODEL + (size_t)h * BATCH * NTOK +
          (size_t)(blk * GB + g) * NTOK + lane] = a;
    }
  }
  __syncthreads();

  // ---- attn @ V -> fcin rows 0..7 (f16); zero rows 8..15 ----
  {
    const int h = t >> 5, dv = t & 31;
    const int col = h * 32 + dv;
    for (int g = 0; g < GB; ++g) {
      float o = 0.f;
#pragma unroll
      for (int n = 0; n < NTOK; ++n)
        o += sc[g * 256 + h * 32 + n] * (float)mat[(g * 32 + n) * R16 + col];
      fcin[g * R16 + col] = (_Float16)o;
    }
    for (int gz = 0; gz < 8; ++gz) fcin[(8 + gz) * R16 + t] = (_Float16)0.0f;
  }
  __syncthreads();

  // ---- FC as WMMA: A rows = batches (type-masked by u), one M-tile ----
  {
    const int myu = us[hl];                 // -1 pads rows 8..15 (also zeroed)
    v8f facc[2] = {};
    for (int e = 0; e < NTYPES; ++e) {
      const _Float16 s = (myu == e) ? (_Float16)1.0f : (_Float16)0.0f;
      for (int ks = 0; ks < 8; ++ks) {
        const int kb = ks * 32 + hi * 8;
        const v16h a = load_afrag(fcin + hl * R16, kb) * s;
#pragma unroll
        for (int j = 0; j < 2; ++j) {
          const int nt = wave * 2 + j;
          const v16h b = *(const v16h*)(fFc + (((e * 16 + nt) * 8 + ks) << 9) + lane * 16);
          facc[j] = __builtin_amdgcn_wmma_f32_16x16x32_f16(
              false, a, false, b, (short)0, facc[j], false, false);
        }
      }
    }
    if (hi == 0) {                          // rows 0..7 = valid batches
#pragma unroll
      for (int j = 0; j < 2; ++j) {
        const int col = (wave * 2 + j) * 16 + hl;
#pragma unroll
        for (int r = 0; r < 8; ++r) proj[r * 256 + col] = facc[j][r];
      }
    }
  }
  __syncthreads();

  // ---- bias + LayerNorm: wave w = batch g=w ----
  {
    const int g = wave;
    const int ug = us[g];
    float vals[8], s1 = 0.f, s2 = 0.f;
#pragma unroll
    for (int j = 0; j < 8; ++j) {
      const int c = j * 32 + lane;
      const float val = proj[g * 256 + c] + fcb[ug * DMODEL + c];
      vals[j] = val; s1 += val; s2 += val * val;
    }
    for (int off = 16; off > 0; off >>= 1) { s1 += __shfl_xor(s1, off); s2 += __shfl_xor(s2, off); }
    const float mu  = s1 * (1.0f / 256.0f);
    const float var = s2 * (1.0f / 256.0f) - mu * mu;
    const float rs  = rsqrtf(var + 1e-5f);
#pragma unroll
    for (int j = 0; j < 8; ++j) {
      const int c = j * 32 + lane;
      out[(size_t)(blk * GB + g) * DMODEL + c] = (vals[j] - mu) * rs * gamma_[c] + beta_[c];
    }
  }
}

// ---------------------------------------------------------------------------
extern "C" void kernel_launch(void* const* d_in, const int* in_sizes, int n_in,
                              void* d_out, int out_size, void* d_ws, size_t ws_size,
                              hipStream_t stream) {
  const float* q     = (const float*)d_in[0];
  const float* k     = (const float*)d_in[1];
  const float* v     = (const float*)d_in[2];
  const float* Wq    = (const float*)d_in[3];
  const float* Wk    = (const float*)d_in[4];
  const float* Wv    = (const float*)d_in[5];
  const float* rel   = (const float*)d_in[6];
  const float* fcw   = (const float*)d_in[7];
  const float* fcb   = (const float*)d_in[8];
  const float* gam   = (const float*)d_in[9];
  const float* bet   = (const float*)d_in[10];
  const int*   etype = (const int*)d_in[11];
  const int*   utype = (const int*)d_in[12];
  const unsigned char* mask = (const unsigned char*)d_in[13];  // jnp bool = 1 byte

  _Float16* fWq = (_Float16*)d_ws;
  _Float16* fWk = fWq + FRAG_HALVES;
  _Float16* fWv = fWk + FRAG_HALVES;
  _Float16* fFc = fWv + FRAG_HALVES;

  pack_w_kernel<<<FRAG_HALVES / 256, 256, 0, stream>>>(Wq, fWq);
  pack_w_kernel<<<FRAG_HALVES / 256, 256, 0, stream>>>(Wk, fWk);
  pack_w_kernel<<<FRAG_HALVES / 256, 256, 0, stream>>>(Wv, fWv);
  pack_w_kernel<<<FRAG_HALVES / 256, 256, 0, stream>>>(fcw, fFc);

  mha_fused_kernel<<<NBLK, 256, SMEM_BYTES, stream>>>(
      q, k, v, fWq, fWk, fWv, fFc, rel, fcb, gam, bet, etype, utype, mask,
      (float*)d_out);

  (void)in_sizes; (void)n_in; (void)out_size; (void)ws_size;
}
